// TAdaConv2D_CinAda_AV_42039139893847
// MI455X (gfx1250) — matrix operand order repaired and checked by reading, not compiled, not run on hardware
//
#include <hip/hip_runtime.h>

typedef __attribute__((ext_vector_type(16))) __bf16 v16bf;
typedef __attribute__((ext_vector_type(8)))  float  v8f;

__device__ __forceinline__ unsigned short f2bf(float f) {
    unsigned int u = __float_as_uint(f);
    u += 0x7FFFu + ((u >> 16) & 1u);      // round-to-nearest-even
    return (unsigned short)(u >> 16);
}

// x:   [16, 64, 8, 56, 56]  fp32
// rw:  [16, 64, 8, 1, 1]    fp32
// w:   [64, 64, 3, 3]       fp32
// out: [16, 64, 8, 56, 56]  fp32
__global__ __launch_bounds__(256)
void TAdaConv2D_CinAda_AV_wmma_kernel(const float* __restrict__ x,
                                      const float* __restrict__ rw,
                                      const float* __restrict__ w,
                                      float* __restrict__ out) {
    constexpr int CIN = 64, T = 8, HW = 56 * 56, COUT = 64;

    // A-fragments: (tap, mTile) blocks of 32 lanes x 16 bf16 (contiguous per lane)
    __shared__ __align__(64) unsigned short ldsW[9 * 4 * 32 * 16];   // 36864 B
    // x tile: [6 rows][60 cols][32 ch] bf16, halo zero-padded       // 23040 B
    __shared__ __align__(64) unsigned short ldsX[6 * 60 * 32];

    const int tid   = threadIdx.x;
    const int lane  = tid & 31;
    const int wave  = tid >> 5;
    const int mTile = wave & 3;       // which 16 output channels
    const int nHalf = wave >> 2;      // which 112-pixel half
    const int h16   = lane >> 4;      // half-wave id (K-half / M-half)
    const int nloc  = lane & 15;      // column within 16x16 tile

    const int yq  = blockIdx.x;       // 0..13 : row quad
    const int img = blockIdx.y;       // 0..127: (b,t)
    const int bI  = img >> 3;
    const int tI  = img & 7;
    const int y0  = yq * 4;

    const float* xImg = x + ((size_t)bI * CIN * T + tI) * HW;  // +c*T*HW later

    v8f acc[7];
    v8f zero = {0.f, 0.f, 0.f, 0.f, 0.f, 0.f, 0.f, 0.f};
#pragma unroll
    for (int i = 0; i < 7; ++i) acc[i] = zero;

    // Per-nt pixel geometry (constant across K loop)
    int rowcol[7];   // yl*60 + (xc+?) base in padded LDS coords
    int pixArr[7];
#pragma unroll
    for (int nt = 0; nt < 7; ++nt) {
        int pix = (nHalf * 7 + nt) * 16 + nloc;   // 0..223
        int yl  = pix / 56;                       // 0..3
        int xc  = pix - yl * 56;                  // 0..55
        rowcol[nt] = yl * 60 + xc;                // tap adds dy*60 + dx
        pixArr[nt] = pix;
    }

#pragma unroll
    for (int chalf = 0; chalf < 2; ++chalf) {
        const int c0 = chalf * 32;
        __syncthreads();   // protect LDS before restaging

        // ---- stage swizzled bf16 weights: A-fragment layout ----
        for (int e = tid; e < 9 * 4 * 32 * 16; e += 256) {
            int j    = e & 15;
            int slot = e >> 4;
            int ln   = slot & 31;
            int mt   = (slot >> 5) & 3;
            int tap  = slot >> 7;
            int hh   = ln >> 4;
            int M    = mt * 16 + (ln & 15);
            // A k order used here: lanes<16: K=0..15; lanes>=16: K=16..31 (contiguous, matches B)
            int k    = j + 16 * hh;
            ldsW[e]  = f2bf(w[(M * CIN + (c0 + k)) * 9 + tap]);
        }

        // ---- stage x tile: scale by routing, convert to bf16, ch-interleaved ----
        for (int f = tid; f < 32 * 6 * 56; f += 256) {
            int xc  = f % 56;
            int kr  = f / 56;
            int r   = kr % 6;          // LDS row 0..5 -> y_in = y0-1+r
            int k   = kr / 6;          // channel within half
            int c   = c0 + k;
            int yin = y0 - 1 + r;
            float v = 0.f;
            if ((unsigned)yin < 56u)
                v = xImg[(size_t)c * T * HW + yin * 56 + xc] *
                    rw[(bI * CIN + c) * T + tI];
            ldsX[(r * 60 + (xc + 1)) * 32 + k] = f2bf(v);
        }
        // zero halo columns 0 and 57
        for (int f = tid; f < 6 * 2 * 32; f += 256) {
            int k   = f & 31;
            int q   = f >> 5;              // 0..11
            int r   = (q < 6) ? q : q - 6;
            int col = (q < 6) ? 0 : 57;
            ldsX[(r * 60 + col) * 32 + k] = 0;
        }
        __syncthreads();

        // ---- K loop: 9 taps x K=32 ----
#pragma unroll
        for (int tap = 0; tap < 9; ++tap) {
            const int dy = tap / 3, dx = tap % 3;
            const v16bf aF = *reinterpret_cast<const v16bf*>(
                &ldsW[((tap * 4 + mTile) * 32 + lane) * 16]);
#pragma unroll
            for (int nt = 0; nt < 7; ++nt) {
                // B k order used here: lane = column N, K = 16 contiguous (16*h16 ..)
                const v16bf bF = *reinterpret_cast<const v16bf*>(
                    &ldsX[(rowcol[nt] + dy * 60 + dx) * 32 + 16 * h16]);
                acc[nt] = __builtin_amdgcn_wmma_f32_16x16x32_bf16(
                    false, aF, false, bF, (short)0, acc[nt], false, false);
            }
        }
    }

    // ---- write f32 output: C/D layout (VGPR v, lane L) -> M=v+8*h16, N=L%16 ----
    const int pixBase = yq * 224;
#pragma unroll
    for (int nt = 0; nt < 7; ++nt) {
        const size_t pix = (size_t)pixBase + pixArr[nt];
#pragma unroll
        for (int v = 0; v < 8; ++v) {
            int cOut = mTile * 16 + v + 8 * h16;
            out[(((size_t)bI * COUT + cOut) * T + tI) * HW + pix] = acc[nt][v];
        }
    }
}

extern "C" void kernel_launch(void* const* d_in, const int* in_sizes, int n_in,
                              void* d_out, int out_size, void* d_ws, size_t ws_size,
                              hipStream_t stream) {
    (void)in_sizes; (void)n_in; (void)d_ws; (void)ws_size; (void)out_size;
    const float* x  = (const float*)d_in[0];
    const float* rw = (const float*)d_in[1];
    const float* w  = (const float*)d_in[2];
    float* out      = (float*)d_out;
    dim3 grid(14, 128);   // 14 row-quads x 128 (b,t) images
    TAdaConv2D_CinAda_AV_wmma_kernel<<<grid, 256, 0, stream>>>(x, rw, w, out);
}
